// DenseInterQTripletLoss_24550033064525
// MI455X (gfx1250) — matrix operand order repaired and checked by reading, not compiled, not run on hardware
//
#include <hip/hip_runtime.h>

#define GS 8
#define BIGV 5.0f
#define HC 64
#define WC 64
#define FLAT 4096
#define CDIM 256
#define HH 512
#define WW 512
#define NB 2
#define BROW 264   // padded LDS row stride in halves (528 B -> conflict-free banks)

typedef __attribute__((ext_vector_type(16))) _Float16 v16h;
typedef __attribute__((ext_vector_type(8)))  _Float16 v8h;
typedef __attribute__((ext_vector_type(8)))  float    v8f;

// ---- ordered-uint encoding of float (monotone for atomicMin over all floats) ----
__device__ __forceinline__ unsigned f2ord(float f) {
    unsigned u = __float_as_uint(f);
    return (u & 0x80000000u) ? ~u : (u | 0x80000000u);
}
__device__ __forceinline__ float ord2f(unsigned u) {
    return __uint_as_float((u & 0x80000000u) ? (u ^ 0x80000000u) : ~u);
}

// ============ K0: (b,c,flat) f32 -> (b,flat,c) f16 ============
__global__ __launch_bounds__(256) void convert_kernel(const float* __restrict__ src,
                                                      _Float16* __restrict__ dst) {
    size_t idx = (size_t)blockIdx.x * 256 + threadIdx.x;   // over b*c*flat, n fastest
    int n = (int)(idx & (FLAT - 1));
    size_t bk = idx >> 12;            // b*CDIM + k
    int k = (int)(bk & (CDIM - 1));
    int b = (int)(bk >> 8);
    dst[((size_t)b * FLAT + n) * CDIM + k] = (_Float16)src[idx];
}

// ============ K1: per-point warp, bilinear dot (pos), wv, neighbor ids ============
__global__ __launch_bounds__(256) void point_kernel(const float* __restrict__ desc1,
                                                    const float* __restrict__ desc2,
                                                    const float* __restrict__ homo,
                                                    float* __restrict__ pos,
                                                    float* __restrict__ wv,
                                                    int4* __restrict__ ids) {
    int gwave = (int)((blockIdx.x * 256 + threadIdx.x) >> 5);  // one wave per (b,n)
    int lane  = threadIdx.x & 31;
    int b = gwave >> 12;
    int n = gwave & (FLAT - 1);
    int gy = n >> 6, gx = n & 63;

    const float* Hm = homo + b * 9;
    float X = (float)(gx * GS), Y = (float)(gy * GS);
    float w0 = Hm[0] * X + Hm[1] * Y + Hm[2];
    float w1 = Hm[3] * X + Hm[4] * Y + Hm[5];
    float w2 = Hm[6] * X + Hm[7] * Y + Hm[8];
    float invw = 1.0f / (w2 + 1e-8f);
    float Xw = w0 * invw;                 // w_coo[...,1] (x)
    float Yw = w1 * invw;                 // w_coo[...,0] (y) after the ::-1 flip
    float wvv = (Yw >= 0.0f && Yw < (float)HH && Xw >= 0.0f && Xw < (float)WW) ? 1.0f : 0.0f;

    // bilinear in (hc,wc) grid at coords /GS
    float yc = fminf(fmaxf(Yw * 0.125f, 0.0f), 63.0f);
    float xc = fminf(fmaxf(Xw * 0.125f, 0.0f), 63.0f);
    float y0 = floorf(yc), x0 = floorf(xc);
    float y1 = fminf(y0 + 1.0f, 63.0f), x1 = fminf(x0 + 1.0f, 63.0f);
    float wy = yc - y0, wx = xc - x0;
    int i00 = (int)y0 * WC + (int)x0;
    int i01 = (int)y0 * WC + (int)x1;
    int i10 = (int)y1 * WC + (int)x0;
    int i11 = (int)y1 * WC + (int)x1;
    float a00 = (1.0f - wy) * (1.0f - wx), a01 = (1.0f - wy) * wx;
    float a10 = wy * (1.0f - wx),          a11 = wy * wx;

    const float* D1 = desc1 + (size_t)b * CDIM * FLAT;
    const float* D2 = desc2 + (size_t)b * CDIM * FLAT;
    float dot = 0.0f;
    for (int k = lane; k < CDIM; k += 32) {
        const float* Dk = D2 + (size_t)k * FLAT;
        float v = Dk[i00] * a00 + Dk[i01] * a01 + Dk[i10] * a10 + Dk[i11] * a11;
        dot += D1[(size_t)k * FLAT + n] * v;
    }
    for (int off = 16; off; off >>= 1) dot += __shfl_xor(dot, off, 32);

    if (lane == 0) {
        pos[b * FLAT + n] = 2.0f - 2.0f * dot;
        wv[b * FLAT + n]  = wvv;
        // argmin over grid centers (gy2*8+4, gx2*8+4); ties -> lower index
        int gy2 = min(max((int)ceilf((Yw - 4.0f) * 0.125f - 0.5f), 0), 63);
        int gx2 = min(max((int)ceilf((Xw - 4.0f) * 0.125f - 0.5f), 0), 63);
        int ul = gy2 * WC + gx2;
        int ur = (ul + 1  >= FLAT) ? ul : ul + 1;
        int ll = (ul + WC >= FLAT) ? ul : ul + WC;
        int lr = (ll + 1  >= FLAT) ? ll : ll + 1;
        ids[b * FLAT + n] = make_int4(ul, ur, ll, lr);
    }
}

// ============ K2: 8x8 block AND of bool mask -> column bias BIG*(1-vis) ============
__global__ __launch_bounds__(256) void vis_kernel(const unsigned char* __restrict__ mask,
                                                  float* __restrict__ colbias) {
    int idx = blockIdx.x * 256 + threadIdx.x;   // b*FLAT
    if (idx >= NB * FLAT) return;
    int b = idx >> 12, m = idx & (FLAT - 1);
    int gy = m >> 6, gx = m & 63;
    const unsigned char* base = mask + ((size_t)b * HH + (size_t)gy * GS) * WW + (size_t)gx * GS;
    bool all = true;
    #pragma unroll
    for (int r = 0; r < GS; r++) {
        uint2 v = *(const uint2*)(base + (size_t)r * WW);   // 8 bytes, 8B aligned
        unsigned za = (v.x - 0x01010101u) & ~v.x & 0x80808080u;  // any zero byte?
        unsigned zb = (v.y - 0x01010101u) & ~v.y & 0x80808080u;
        if (za | zb) { all = false; }
    }
    colbias[idx] = all ? 0.0f : BIGV;
}

// ============ K3: init neg keys ============
__global__ __launch_bounds__(256) void init_kernel(unsigned* __restrict__ negbits) {
    int idx = blockIdx.x * 256 + threadIdx.x;
    if (idx < NB * FLAT) negbits[idx] = 0xFFFFFFFFu;
}

// ============ K4: WMMA GEMM, B tile staged in LDS via async-to-LDS, ============
// ============ A+B register double-buffered, fused min epilogue       ============
__global__ __launch_bounds__(256) void gemm_min_kernel(const _Float16* __restrict__ d1h,
                                                       const _Float16* __restrict__ d2h,
                                                       const float* __restrict__ colbias,
                                                       const int4* __restrict__ ids,
                                                       unsigned* __restrict__ negbits) {
    __shared__ _Float16 Bt[64 * BROW];                  // 64 rows x 264 halves = 33 KB

    const int b     = blockIdx.z;
    const int wave  = threadIdx.x >> 5;
    const int lane  = threadIdx.x & 31;
    const int laneM = lane & 15;
    const int hi    = lane >> 4;
    const int rowBase = blockIdx.x * 128 + wave * 16;   // 8 waves x 16 rows
    const int colBase = blockIdx.y * 64;                // 4 tiles of 16 cols

    // ---- stage B tile (64 x 256 f16) into LDS with CDNA5 async-to-LDS copies ----
    // 2048 x 16B chunks; 256 threads -> 8 chunks each. Row padded to BROW halves.
    {
        const _Float16* Bsrc = d2h + ((size_t)b * FLAT + colBase) * CDIM;
        #pragma unroll
        for (int i = 0; i < 8; i++) {
            int chunk = threadIdx.x + i * 256;          // [0, 2048)
            int row   = chunk >> 5;                     // 32 chunks per row
            int seg   = chunk & 31;
            unsigned long long gaddr =
                (unsigned long long)(uintptr_t)(Bsrc + (size_t)row * CDIM + seg * 8);
            unsigned lds_off =
                (unsigned)(uintptr_t)(&Bt[row * BROW + seg * 8]);
            asm volatile("global_load_async_to_lds_b128 %0, %1, off"
                         :: "v"(lds_off), "v"(gaddr) : "memory");
        }
        asm volatile("s_wait_asynccnt 0x0" ::: "memory");
        __syncthreads();
    }

    const _Float16* Arow = d1h + ((size_t)b * FLAT + rowBase + laneM) * CDIM;
    const _Float16* L0 = &Bt[( 0 + laneM) * BROW + hi * 16];
    const _Float16* L1 = &Bt[(16 + laneM) * BROW + hi * 16];
    const _Float16* L2 = &Bt[(32 + laneM) * BROW + hi * 16];
    const _Float16* L3 = &Bt[(48 + laneM) * BROW + hi * 16];

    v8f acc0 = {}, acc1 = {}, acc2 = {}, acc3 = {};

    // Software pipeline: fragments for step k pre-loaded during step k-1.
    v8h alo = *(const v8h*)(Arow + hi * 16);
    v8h ahi = *(const v8h*)(Arow + hi * 16 + 8);
    v16h b0c = *(const v16h*)(L0);
    v16h b1c = *(const v16h*)(L1);
    v16h b2c = *(const v16h*)(L2);
    v16h b3c = *(const v16h*)(L3);

    #pragma unroll
    for (int k = 0; k < CDIM; k += 32) {
        v8h  nlo = alo, nhi = ahi;
        v16h b0n = b0c, b1n = b1c, b2n = b2c, b3n = b3c;
        if (k + 32 < CDIM) {
            nlo = *(const v8h*)(Arow + (k + 32) + hi * 16);
            nhi = *(const v8h*)(Arow + (k + 32) + hi * 16 + 8);
            b0n = *(const v16h*)(L0 + (k + 32));
            b1n = *(const v16h*)(L1 + (k + 32));
            b2n = *(const v16h*)(L2 + (k + 32));
            b3n = *(const v16h*)(L3 + (k + 32));
        }
        v16h a = __builtin_shufflevector(alo, ahi, 0,1,2,3,4,5,6,7,8,9,10,11,12,13,14,15);
        acc0 = __builtin_amdgcn_wmma_f32_16x16x32_f16(false, a, false, b0c, (short)0, acc0, false, false);
        acc1 = __builtin_amdgcn_wmma_f32_16x16x32_f16(false, a, false, b1c, (short)0, acc1, false, false);
        acc2 = __builtin_amdgcn_wmma_f32_16x16x32_f16(false, a, false, b2c, (short)0, acc2, false, false);
        acc3 = __builtin_amdgcn_wmma_f32_16x16x32_f16(false, a, false, b3c, (short)0, acc3, false, false);
        alo = nlo; ahi = nhi;
        b0c = b0n; b1c = b1n; b2c = b2n; b3c = b3n;
    }

    // Epilogue: sim = 2 - 2*dot + colbias[N] + BIG*[N in ids(M)], then row-min.
    const int N0 = colBase + laneM, N1 = N0 + 16, N2 = N0 + 32, N3 = N0 + 48;
    const float cb0 = colbias[b * FLAT + N0];
    const float cb1 = colbias[b * FLAT + N1];
    const float cb2 = colbias[b * FLAT + N2];
    const float cb3 = colbias[b * FLAT + N3];

    float rmin[8];
    #pragma unroll
    for (int r = 0; r < 8; r++) {
        int Mg = rowBase + r + 8 * hi;          // C/D layout: lanes 0-15 M=r, 16-31 M=r+8
        int4 idv = ids[b * FLAT + Mg];
        float s0 = 2.0f - 2.0f * acc0[r] + cb0;
        float s1 = 2.0f - 2.0f * acc1[r] + cb1;
        float s2 = 2.0f - 2.0f * acc2[r] + cb2;
        float s3 = 2.0f - 2.0f * acc3[r] + cb3;
        if (N0 == idv.x || N0 == idv.y || N0 == idv.z || N0 == idv.w) s0 += BIGV;
        if (N1 == idv.x || N1 == idv.y || N1 == idv.z || N1 == idv.w) s1 += BIGV;
        if (N2 == idv.x || N2 == idv.y || N2 == idv.z || N2 == idv.w) s2 += BIGV;
        if (N3 == idv.x || N3 == idv.y || N3 == idv.z || N3 == idv.w) s3 += BIGV;
        rmin[r] = fminf(fminf(s0, s1), fminf(s2, s3));
    }
    // min across the 16 lanes of each half (xor 1,2,4,8 keeps halves separate)
    #pragma unroll
    for (int off = 1; off < 16; off <<= 1) {
        #pragma unroll
        for (int r = 0; r < 8; r++)
            rmin[r] = fminf(rmin[r], __shfl_xor(rmin[r], off, 32));
    }
    if (laneM == 0) {
        #pragma unroll
        for (int r = 0; r < 8; r++)
            atomicMin(negbits + b * FLAT + rowBase + r + 8 * hi, f2ord(rmin[r]));
    }
}

// ============ K5/K6: deterministic loss reduction ============
__global__ __launch_bounds__(256) void loss_partial(const float* __restrict__ pos,
                                                    const unsigned* __restrict__ negbits,
                                                    const float* __restrict__ wv,
                                                    float* __restrict__ partial) {
    __shared__ float sl[256], sw[256];
    int tid = threadIdx.x;
    int idx = blockIdx.x * 256 + tid;
    float l = 0.0f, w = 0.0f;
    if (idx < NB * FLAT) {
        float neg = ord2f(negbits[idx]);
        float wvv = wv[idx];
        float d = fmaxf(pos[idx] - neg + 1.0f, 0.0f);   // MARGIN = 1
        l = d * d * wvv;
        w = wvv;
    }
    sl[tid] = l; sw[tid] = w;
    __syncthreads();
    for (int s = 128; s; s >>= 1) {
        if (tid < s) { sl[tid] += sl[tid + s]; sw[tid] += sw[tid + s]; }
        __syncthreads();
    }
    if (tid == 0) { partial[blockIdx.x] = sl[0]; partial[gridDim.x + blockIdx.x] = sw[0]; }
}

__global__ void loss_final(const float* __restrict__ partial, int nb, float* __restrict__ out) {
    if (threadIdx.x == 0 && blockIdx.x == 0) {
        float l = 0.0f, w = 0.0f;
        for (int i = 0; i < nb; i++) { l += partial[i]; w += partial[nb + i]; }
        out[0] = l / w * 1.0f;   // LAMBDA = 1
    }
}

extern "C" void kernel_launch(void* const* d_in, const int* in_sizes, int n_in,
                              void* d_out, int out_size, void* d_ws, size_t ws_size,
                              hipStream_t stream) {
    const float*         desc1 = (const float*)d_in[0];   // (2,256,64,64) f32
    const float*         desc2 = (const float*)d_in[1];   // (2,256,64,64) f32
    const float*         homo  = (const float*)d_in[2];   // (2,3,3) f32
    const unsigned char* vmask = (const unsigned char*)d_in[3]; // (2,1,512,512) bool
    // d_in[4] = score2 : unused (shape only)
    float* out = (float*)d_out;

    // ---- workspace carve ----
    char* p = (char*)d_ws;
    _Float16* d1h = (_Float16*)p;  p += (size_t)NB * FLAT * CDIM * sizeof(_Float16); // 4 MB
    _Float16* d2h = (_Float16*)p;  p += (size_t)NB * FLAT * CDIM * sizeof(_Float16); // 4 MB
    float*   colbias = (float*)p;  p += (size_t)NB * FLAT * sizeof(float);
    int4*    ids     = (int4*)p;   p += (size_t)NB * FLAT * sizeof(int4);
    float*   pos     = (float*)p;  p += (size_t)NB * FLAT * sizeof(float);
    float*   wv      = (float*)p;  p += (size_t)NB * FLAT * sizeof(float);
    unsigned* negb   = (unsigned*)p; p += (size_t)NB * FLAT * sizeof(unsigned);
    float*   partial = (float*)p;  p += 2 * 32 * sizeof(float);

    const size_t nconv = (size_t)NB * CDIM * FLAT;          // 2,097,152
    convert_kernel<<<dim3((unsigned)(nconv / 256)), dim3(256), 0, stream>>>(desc1, d1h);
    convert_kernel<<<dim3((unsigned)(nconv / 256)), dim3(256), 0, stream>>>(desc2, d2h);

    point_kernel<<<dim3(NB * FLAT / 8), dim3(256), 0, stream>>>(desc1, desc2, homo, pos, wv, ids);
    vis_kernel<<<dim3(NB * FLAT / 256), dim3(256), 0, stream>>>(vmask, colbias);
    init_kernel<<<dim3(NB * FLAT / 256), dim3(256), 0, stream>>>(negb);

    // GEMM: block = 128 rows x 64 cols; grid = (4096/128, 4096/64, b)
    gemm_min_kernel<<<dim3(FLAT / 128, FLAT / 64, NB), dim3(256), 0, stream>>>(
        d1h, d2h, colbias, ids, negb);

    loss_partial<<<dim3(NB * FLAT / 256), dim3(256), 0, stream>>>(pos, negb, wv, partial);
    loss_final<<<dim3(1), dim3(32), 0, stream>>>(partial, NB * FLAT / 256, out);
}
